// xLSTM_Model_38689065402947
// MI455X (gfx1250) — matrix operand order repaired and checked by reading, not compiled, not run on hardware
//
#include <hip/hip_runtime.h>
#include <math.h>

typedef __bf16 bf16;
typedef __attribute__((ext_vector_type(16))) __bf16 v16bf;
typedef __attribute__((ext_vector_type(8)))  __bf16 v8bf;
typedef __attribute__((ext_vector_type(8)))  float  v8f;

#define B_   64
#define T_   2048
#define P_   16
#define H1_  200
#define G1_  800            // 4*H1
#define BT_  (B_ * T_)      // 131072
#define ASTRIDE 416         // max padded K (layer2: 200+200 -> 416), multiple of 8

__device__ __forceinline__ float gelu_exact(float x) {
    return 0.5f * x * (1.0f + erff(x * 0.70710678118654752440f));
}
__device__ __forceinline__ float sigmoidf_(float x) {
    return 1.0f / (1.0f + expf(-x));
}

// Pack [W | R | zero-pad] (f32) into a KP-padded bf16 matrix, rows padded to `rows`.
__global__ void pack_kernel(bf16* __restrict__ dst,
                            const float* __restrict__ W, int wc,
                            const float* __restrict__ R, int rc,
                            int rows, int realrows, int KP)
{
    int idx = blockIdx.x * blockDim.x + threadIdx.x;
    if (idx >= rows * KP) return;
    int r = idx / KP, k = idx % KP;
    float v = 0.0f;
    if (r < realrows) {
        if (k < wc)            v = W[r * wc + k];
        else if (k < wc + rc)  v = R[r * rc + (k - wc)];
    }
    dst[idx] = (bf16)v;
}

// Persistent sLSTM scan: one block handles 16 batch rows; per step computes
// g = [x_t | h_prev] @ WB^T + b with v_wmma_f32_16x16x32_bf16, then gates in f32.
__global__ __launch_bounds__(512)
void slstm_scan_kernel(const bf16* __restrict__ WB, const float* __restrict__ bvec,
                       const float* __restrict__ disp, const float* __restrict__ params,
                       const bf16* __restrict__ h_in, bf16* __restrict__ h_out,
                       int KP, int x_cols, int mode)
{
    __shared__ bf16  Abuf[16 * ASTRIDE];   // [m][k] : x_t region then h_prev region
    __shared__ float gbuf[16 * G1_];       // pre-activations [m][4H]

    const int tid   = threadIdx.x;
    const int wid   = tid >> 5;
    const int lane  = tid & 31;
    const int b0    = blockIdx.x * 16;
    const int KT    = KP >> 5;
    const int hoff  = x_cols;
    const int nl    = lane & 15;
    const int khalf = lane >> 4;           // 0 or 1

    for (int i = tid; i < 16 * ASTRIDE; i += 512) Abuf[i] = (bf16)0.0f;
    __syncthreads();
    if (mode == 0 && tid < 256) {          // static parameter columns 1..16
        int m = tid >> 4, p = tid & 15;
        Abuf[m * ASTRIDE + 1 + p] = (bf16)params[(b0 + m) * P_ + p];
    }
    __syncthreads();

    // per-thread recurrent state cells: cell idx = tid + 512*j -> (m = idx/200, u = idx%200)
    float cS[7], nS[7], mS[7];
#pragma unroll
    for (int j = 0; j < 7; ++j) { cS[j] = 0.0f; nS[j] = 0.0f; mS[j] = 0.0f; }

    for (int t = 0; t < T_; ++t) {
        // ---- fill dynamic input region of A ----
        if (mode == 0) {
            if (tid < 16)
                Abuf[tid * ASTRIDE + 0] = (bf16)disp[(size_t)(b0 + tid) * T_ + t];
        } else {
            for (int idx = tid; idx < 16 * H1_; idx += 512) {
                int m = idx / H1_, u = idx % H1_;
                Abuf[m * ASTRIDE + u] = h_in[((size_t)(b0 + m) * T_ + t) * H1_ + u];
            }
        }
        __syncthreads();

        // ---- WMMA: each wave covers n-tiles wid, wid+16, ... of 50 ----
        for (int nt = wid; nt < 50; nt += 16) {
            int ng = nt * 16 + nl;
            float bias = bvec[ng];
            v8f acc = { bias, bias, bias, bias, bias, bias, bias, bias };
            const bf16* brow = WB + (size_t)ng * KP;
            const bf16* arow = Abuf + nl * ASTRIDE;
            for (int kt = 0; kt < KT; ++kt) {
                int kbase = kt * 32 + khalf * 8;
                v8bf alo = *(const v8bf*)(arow + kbase);
                v8bf ahi = *(const v8bf*)(arow + kbase + 16);
                v16bf a;
#pragma unroll
                for (int i = 0; i < 8; ++i) { a[i] = alo[i]; a[8 + i] = ahi[i]; }
                v8bf blo = *(const v8bf*)(brow + kbase);
                v8bf bhi = *(const v8bf*)(brow + kbase + 16);
                v16bf b;
#pragma unroll
                for (int i = 0; i < 8; ++i) { b[i] = blo[i]; b[8 + i] = bhi[i]; }
                acc = __builtin_amdgcn_wmma_f32_16x16x32_bf16(
                          false, a, false, b, (short)0, acc, false, false);
            }
            int mb = khalf * 8;
#pragma unroll
            for (int r = 0; r < 8; ++r)
                gbuf[(mb + r) * G1_ + nt * 16 + nl] = acc[r];
        }
        __syncthreads();

        // ---- stabilized sLSTM gates (f32) ----
#pragma unroll
        for (int j = 0; j < 7; ++j) {
            int idx = tid + j * 512;
            if (idx < 16 * H1_) {
                int m = idx / H1_, u = idx % H1_;
                float it = gbuf[m * G1_ + u];
                float ft = gbuf[m * G1_ + H1_ + u];
                float zt = gbuf[m * G1_ + 2 * H1_ + u];
                float ot = gbuf[m * G1_ + 3 * H1_ + u];
                float mn = fmaxf(ft + mS[j], it);
                float iv = expf(it - mn);
                float fv = expf(ft + mS[j] - mn);
                float cv = fv * cS[j] + iv * tanhf(zt);
                float nv = fv * nS[j] + iv;
                float hv = sigmoidf_(ot) * (cv / nv);
                cS[j] = cv; nS[j] = nv; mS[j] = mn;
                bf16 hb = (bf16)hv;
                Abuf[m * ASTRIDE + hoff + u] = hb;
                h_out[((size_t)(b0 + m) * T_ + t) * H1_ + u] = hb;
            }
        }
        __syncthreads();
    }
}

// Decoder: both H=1 sLSTM layers fused; one wave per batch row.
__global__ __launch_bounds__(512)
void dec_scan_kernel(const bf16* __restrict__ h2,
                     const float* __restrict__ Wd1, const float* __restrict__ Rd1,
                     const float* __restrict__ bd1, const float* __restrict__ Wd2,
                     const float* __restrict__ Rd2, const float* __restrict__ bd2,
                     float* __restrict__ hd)
{
    const int lane = threadIdx.x & 31;
    const int wid  = threadIdx.x >> 5;
    const int b    = blockIdx.x * 16 + wid;

    float w1r[4][7];
#pragma unroll
    for (int g = 0; g < 4; ++g)
#pragma unroll
        for (int j = 0; j < 7; ++j) {
            int u = lane + 32 * j;
            w1r[g][j] = (u < H1_) ? Wd1[g * H1_ + u] : 0.0f;
        }
    float r1[4], bb1[4], w2[4], r2[4], bb2[4];
#pragma unroll
    for (int g = 0; g < 4; ++g) {
        r1[g] = Rd1[g]; bb1[g] = bd1[g];
        w2[g] = Wd2[g]; r2[g] = Rd2[g]; bb2[g] = bd2[g];
    }

    float h1s = 0, c1 = 0, n1 = 0, m1 = 0;
    float h2s = 0, c2 = 0, n2 = 0, m2 = 0;

    for (int t = 0; t < T_; ++t) {
        const bf16* xr = h2 + ((size_t)b * T_ + t) * H1_;
        float p[4] = {0, 0, 0, 0};
#pragma unroll
        for (int j = 0; j < 7; ++j) {
            int u = lane + 32 * j;
            float xv = (u < H1_) ? gelu_exact((float)xr[u]) : 0.0f;
#pragma unroll
            for (int g = 0; g < 4; ++g) p[g] += xv * w1r[g][j];
        }
#pragma unroll
        for (int off = 16; off > 0; off >>= 1)
#pragma unroll
            for (int g = 0; g < 4; ++g) p[g] += __shfl_xor(p[g], off, 32);

        {   // layer 1 (200 -> 1)
            float it = p[0] + bb1[0] + r1[0] * h1s;
            float ft = p[1] + bb1[1] + r1[1] * h1s;
            float zt = p[2] + bb1[2] + r1[2] * h1s;
            float ot = p[3] + bb1[3] + r1[3] * h1s;
            float mn = fmaxf(ft + m1, it);
            float iv = expf(it - mn), fv = expf(ft + m1 - mn);
            c1 = fv * c1 + iv * tanhf(zt);
            n1 = fv * n1 + iv;
            h1s = sigmoidf_(ot) * (c1 / n1);
            m1 = mn;
        }
        {   // layer 2 (1 -> 1)
            float it = w2[0] * h1s + bb2[0] + r2[0] * h2s;
            float ft = w2[1] * h1s + bb2[1] + r2[1] * h2s;
            float zt = w2[2] * h1s + bb2[2] + r2[2] * h2s;
            float ot = w2[3] * h1s + bb2[3] + r2[3] * h2s;
            float mn = fmaxf(ft + m2, it);
            float iv = expf(it - mn), fv = expf(ft + m2 - mn);
            c2 = fv * c2 + iv * tanhf(zt);
            n2 = fv * n2 + iv;
            h2s = sigmoidf_(ot) * (c2 / n2);
            m2 = mn;
        }
        if (lane == 0) hd[(size_t)b * T_ + t] = gelu_exact(h2s);
    }
}

// t1[m][u] = tanh(d1_w[u]*hd[m] + d1_b[u]), K-padded to 224, bf16.
__global__ void t1_kernel(const float* __restrict__ hd, const float* __restrict__ d1w,
                          const float* __restrict__ d1b, bf16* __restrict__ T1)
{
    size_t idx = (size_t)blockIdx.x * blockDim.x + threadIdx.x;
    if (idx >= (size_t)BT_ * 224) return;
    int u = (int)(idx % 224);
    size_t m = idx / 224;
    float v = 0.0f;
    if (u < H1_) v = tanhf(d1w[u] * hd[m] + d1b[u]);
    T1[idx] = (bf16)v;
}

// T2 = tanh(T1 @ d2_w^T + d2_b): M=131072, N=112(pad of 100), K=224 — WMMA.
__global__ __launch_bounds__(256)
void head_gemm_kernel(const bf16* __restrict__ T1, const bf16* __restrict__ WD2,
                      const float* __restrict__ d2b, bf16* __restrict__ T2)
{
    const int lane = threadIdx.x & 31;
    const int wid  = threadIdx.x >> 5;
    const int tile = blockIdx.x * 8 + wid;    // 8192 m-tiles x 7 n-tiles
    const int mt   = tile / 7;
    const int nt   = tile % 7;
    const int nl   = lane & 15;
    const int kh   = lane >> 4;

    int ng = nt * 16 + nl;
    float bias = (ng < 100) ? d2b[ng] : 0.0f;
    v8f acc = { bias, bias, bias, bias, bias, bias, bias, bias };
    const bf16* arow = T1 + ((size_t)mt * 16 + nl) * 224;
    const bf16* brow = WD2 + (size_t)ng * 224;
#pragma unroll
    for (int kt = 0; kt < 7; ++kt) {
        int kbase = kt * 32 + kh * 8;
        v8bf alo = *(const v8bf*)(arow + kbase);
        v8bf ahi = *(const v8bf*)(arow + kbase + 16);
        v16bf a;
#pragma unroll
        for (int i = 0; i < 8; ++i) { a[i] = alo[i]; a[8 + i] = ahi[i]; }
        v8bf blo = *(const v8bf*)(brow + kbase);
        v8bf bhi = *(const v8bf*)(brow + kbase + 16);
        v16bf b;
#pragma unroll
        for (int i = 0; i < 8; ++i) { b[i] = blo[i]; b[8 + i] = bhi[i]; }
        acc = __builtin_amdgcn_wmma_f32_16x16x32_bf16(
                  false, a, false, b, (short)0, acc, false, false);
    }
    int mb = kh * 8;
#pragma unroll
    for (int r = 0; r < 8; ++r) {
        size_t row = (size_t)mt * 16 + mb + r;
        T2[row * 112 + nt * 16 + nl] = (bf16)tanhf(acc[r]);
    }
}

// out[m] = out_b + sum_j out_w[j] * T2[m][j]
__global__ void out_kernel(const bf16* __restrict__ T2, const float* __restrict__ ow,
                           const float* __restrict__ ob, float* __restrict__ out)
{
    int m = blockIdx.x * blockDim.x + threadIdx.x;
    if (m >= BT_) return;
    const bf16* row = T2 + (size_t)m * 112;
    float s = ob[0];
#pragma unroll 4
    for (int j = 0; j < 100; ++j) s += ow[j] * (float)row[j];
    out[m] = s;
}

extern "C" void kernel_launch(void* const* d_in, const int* in_sizes, int n_in,
                              void* d_out, int out_size, void* d_ws, size_t ws_size,
                              hipStream_t stream)
{
    const float* params = (const float*)d_in[0];
    const float* disp   = (const float*)d_in[1];
    const float* W1a = (const float*)d_in[2];   // (800,17)
    const float* W1b = (const float*)d_in[3];   // (800,200)
    const float* R1a = (const float*)d_in[4];   // (800,200)
    const float* R1b = (const float*)d_in[5];   // (800,200)
    const float* b1a = (const float*)d_in[6];
    const float* b1b = (const float*)d_in[7];
    const float* Wd1 = (const float*)d_in[8];   // (4,200)
    const float* Wd2 = (const float*)d_in[9];   // (4,1)
    const float* Rd1 = (const float*)d_in[10];  // (4,1)
    const float* Rd2 = (const float*)d_in[11];  // (4,1)
    const float* bd1 = (const float*)d_in[12];
    const float* bd2 = (const float*)d_in[13];
    const float* d1w = (const float*)d_in[14];  // (200,1)
    const float* d1b = (const float*)d_in[15];
    const float* d2w = (const float*)d_in[16];  // (100,200)
    const float* d2b = (const float*)d_in[17];
    const float* ow  = (const float*)d_in[18];  // (1,100)
    const float* ob  = (const float*)d_in[19];

    char* ws = (char*)d_ws;
    size_t off = 0;
    auto alloc = [&](size_t bytes) -> char* {
        char* p = ws + off;
        off = (off + bytes + 255) & ~(size_t)255;
        return p;
    };
    bf16* WB1 = (bf16*)alloc((size_t)800 * 224 * 2);   // [W1 | R1] layer1, KP=224
    bf16* WB2 = (bf16*)alloc((size_t)800 * 416 * 2);   // [W2 | R2] layer2, KP=416
    bf16* WD2 = (bf16*)alloc((size_t)112 * 224 * 2);   // padded d2_w
    bf16* h1  = (bf16*)alloc((size_t)BT_ * 224 * 2);   // enc layer1 out; reused as T1
    bf16* h2  = (bf16*)alloc((size_t)BT_ * 200 * 2);   // enc layer2 out; reused as T2
    float* hd = (float*)alloc((size_t)BT_ * 4);        // decoder output (gelu'd)

    pack_kernel<<<(800 * 224 + 255) / 256, 256, 0, stream>>>(WB1, W1a, 17,  R1a, 200, 800, 800, 224);
    pack_kernel<<<(800 * 416 + 255) / 256, 256, 0, stream>>>(WB2, W1b, 200, R1b, 200, 800, 800, 416);
    pack_kernel<<<(112 * 224 + 255) / 256, 256, 0, stream>>>(WD2, d2w, 200, nullptr, 0, 112, 100, 224);

    slstm_scan_kernel<<<4, 512, 0, stream>>>(WB1, b1a, disp, params, nullptr, h1, 224, 17, 0);
    slstm_scan_kernel<<<4, 512, 0, stream>>>(WB2, b1b, nullptr, nullptr, h1, h2, 416, 200, 1);

    dec_scan_kernel<<<4, 512, 0, stream>>>(h2, Wd1, Rd1, bd1, Wd2, Rd2, bd2, hd);

    bf16* T1 = h1;
    bf16* T2 = h2;
    size_t t1_total = (size_t)BT_ * 224;
    t1_kernel<<<(unsigned)((t1_total + 255) / 256), 256, 0, stream>>>(hd, d1w, d1b, T1);
    head_gemm_kernel<<<(8192 * 7) / 8, 256, 0, stream>>>(T1, WD2, d2b, T2);
    out_kernel<<<(BT_ + 255) / 256, 256, 0, stream>>>(T2, ow, ob, (float*)d_out);
}
